// AbsolutePositionalCrossAttention_62766652064273
// MI455X (gfx1250) — matrix operand order repaired and checked by reading, not compiled, not run on hardware
//
#include <hip/hip_runtime.h>
#include <cstdint>

// ---------------------------------------------------------------------------
// CDNA5 (gfx1250) fused absolute-positional cross-attention.
//   B=16384, N=4, DIM=512, HEADS=8, DH=64.  Rows = B*N = 65536.
//   One workgroup = 64 rows (16 batches), 512 threads = 16 waves (wave32).
//
//   Phase 0: stage ctx tile into LDS as pre-converted hi/lo bf16 A-fragments.
//   Phase 1: q = x @ Wq + bq  for all 512 cols (bf16x3 WMMA) -> q_sh (f32).
//   Per head h:
//     Phase 2: k_h, v_h = ctx_sh @ Wkv cols   (LDS A-frags, zero conversion)
//     Phase 3: 4x4 masked-softmax attention   -> o_sh
//     Phase 4: out_acc += o_h @ Wo[h*64..+63, :]  (bf16x3 WMMA)
//   Epilogue: out = out_acc + bo.
//
//   LDS: ctx hi/lo 128KB + q_sh 128KB + k/v/o 48KB = 304KB (<= 320KB/WG).
// ---------------------------------------------------------------------------

typedef __bf16 v16bf __attribute__((ext_vector_type(16)));
typedef float  v8f   __attribute__((ext_vector_type(8)));

#define DEV static __device__ __forceinline__

__constant__ constexpr float kScale = 0.044194173824159216f;  // 512^-0.5
__constant__ constexpr float kGamma = 0.4f;

DEV v8f vzero8() {
  v8f z = {0.f, 0.f, 0.f, 0.f, 0.f, 0.f, 0.f, 0.f};
  return z;
}

DEV v8f wmma_bf16(v16bf a, v16bf b, v8f c) {
  // D = A(16x32 bf16) * B(32x16 bf16) + C(16x16 f32)
  return __builtin_amdgcn_wmma_f32_16x16x32_bf16(
      /*neg_a=*/false, a, /*neg_b=*/false, b,
      /*c_mod=*/(short)0, c, /*reuse_a=*/false, /*reuse_b=*/false);
}

// Gather 16 f32 for an A-fragment lane: groups [k0..k0+7] and [k0+16..k0+23].
DEV void load_a16(const float* __restrict__ rowp, int k0, float* av) {
  float4 a0 = *(const float4*)(rowp + k0);
  float4 a1 = *(const float4*)(rowp + k0 + 4);
  float4 a2 = *(const float4*)(rowp + k0 + 16);
  float4 a3 = *(const float4*)(rowp + k0 + 20);
  av[0] = a0.x;  av[1] = a0.y;  av[2]  = a0.z;  av[3]  = a0.w;
  av[4] = a1.x;  av[5] = a1.y;  av[6]  = a1.z;  av[7]  = a1.w;
  av[8] = a2.x;  av[9] = a2.y;  av[10] = a2.z;  av[11] = a2.w;
  av[12] = a3.x; av[13] = a3.y; av[14] = a3.z;  av[15] = a3.w;
}

// f32 -> (bf16 hi, bf16 lo) split for ~f32-accurate bf16x3 WMMA.
DEV void cvt_split16(const float* av, v16bf& hi, v16bf& lo) {
#pragma unroll
  for (int e = 0; e < 16; ++e) {
    float x  = av[e];
    __bf16 h = (__bf16)x;
    hi[e]    = h;
    lo[e]    = (__bf16)(x - (float)h);
  }
}

// ---------------------------------------------------------------------------
// Prep: pre-swizzle a row-major weight W[K][Nout] into WMMA B-fragment order.
// Fragment (kc,nc) = K-chunk of 32 x N-chunk of 16, stored as 2 planes
// (hi then lo) of 512 bf16:  out[frag*1024 + plane*512 + lane*16 + e]
// B-matrix 32x16 layout: lane n (0..15) holds K=0..7,16..23 (e), lane n+16: K=8..15,24..31.
// ---------------------------------------------------------------------------
__global__ void pack_weights_kernel(const float* __restrict__ W, int K, int Nout,
                                    __bf16* __restrict__ outp) {
  int t = blockIdx.x * blockDim.x + threadIdx.x;
  int total = (K / 32) * (Nout / 16) * 512;
  if (t >= total) return;
  int e       = t & 15;
  int lane    = (t >> 4) & 31;
  int fragidx = t >> 9;
  int ncN     = Nout / 16;
  int nc      = fragidx % ncN;
  int kc      = fragidx / ncN;
  int k       = kc * 32 + ((lane & 16) ? 8 : 0) + ((e < 8) ? e : e + 8);
  int n       = nc * 16 + (lane & 15);
  float w     = W[(size_t)k * Nout + n];
  __bf16 hi   = (__bf16)w;
  __bf16 lo   = (__bf16)(w - (float)hi);
  size_t base = (size_t)fragidx * 1024 + lane * 16 + e;
  outp[base]       = hi;
  outp[base + 512] = lo;
}

// ---------------------------------------------------------------------------
// Fused kernel.
// ---------------------------------------------------------------------------
__global__ __launch_bounds__(512) void fused_attn_kernel(
    const float* __restrict__ x, const float* __restrict__ ctx,
    const float* __restrict__ bq, const float* __restrict__ bkv,
    const float* __restrict__ bo,
    const __bf16* __restrict__ wqp,   // Wq  packed: 16 kc x 32 nc
    const __bf16* __restrict__ wkvp,  // Wkv packed: 16 kc x 64 nc
    const __bf16* __restrict__ wop,   // Wo  packed: 16 kc x 32 nc
    float* __restrict__ out) {
  // ctx A-fragments, hi plane then lo plane: [(kc*4 + r)*512 + lane*16 + e]
  __shared__ __bf16 ctx_sh[2 * 16 * 4 * 512];  // 128 KB
  __shared__ float  q_sh[64 * 512];            // 128 KB
  __shared__ float  k_sh[64 * 64];             //  16 KB
  __shared__ float  v_sh[64 * 64];             //  16 KB
  __shared__ float  o_sh[64 * 64];             //  16 KB

  const int tid   = threadIdx.x;
  const int wave  = tid >> 5;
  const int lane  = tid & 31;
  const int r     = wave >> 2;             // row block 0..3 (16 rows each)
  const int c     = wave & 3;              // col block 0..3
  const int ln    = lane & 15;
  const int mbase = (lane >> 4) << 3;      // C-matrix M offset (0 or 8)
  const int kbase = (lane & 16) ? 8 : 0;   // A-matrix K group offset
  const int row0  = blockIdx.x * 64;

  // ---- Phase 0: stage ctx tile as hi/lo bf16 A-fragments in LDS -----------
  {
    const int sl = tid >> 4;                       // fragment lane 0..31
    const int se = tid & 15;                       // fragment element 0..15
    const int sm = sl & 15;                        // A row within block
    const int sk = ((sl & 16) ? 8 : 0) + ((se < 8) ? se : se + 8);
    for (int f = 0; f < 64; ++f) {                 // f = kc*4 + rr
      int kc = f >> 2, rr = f & 3;
      float wv = ctx[(size_t)(row0 + rr * 16 + sm) * 512 + kc * 32 + sk];
      __bf16 h = (__bf16)wv;
      __bf16 l = (__bf16)(wv - (float)h);
      int off  = f * 512 + tid;
      ctx_sh[off]         = h;
      ctx_sh[32768 + off] = l;
    }
  }

  // ---- Phase 1: q = x @ Wq + bq (this wave: rows r*16.., cols c*128..) ----
  {
    v8f acc[8];
#pragma unroll
    for (int j = 0; j < 8; ++j) acc[j] = vzero8();
    const float* xrow = x + (size_t)(row0 + r * 16 + ln) * 512;
    for (int kc = 0; kc < 16; ++kc) {
      float av[16];
      v16bf ah, al;
      load_a16(xrow, kc * 32 + kbase, av);
      cvt_split16(av, ah, al);
#pragma unroll
      for (int j = 0; j < 8; ++j) {
        const __bf16* bp = wqp + (size_t)(kc * 32 + c * 8 + j) * 1024 + lane * 16;
        v16bf bh = *(const v16bf*)bp;
        v16bf bl = *(const v16bf*)(bp + 512);
        acc[j] = wmma_bf16(ah, bh, acc[j]);
        acc[j] = wmma_bf16(al, bh, acc[j]);
        acc[j] = wmma_bf16(ah, bl, acc[j]);
      }
    }
#pragma unroll
    for (int j = 0; j < 8; ++j) {
      int col    = c * 128 + j * 16 + ln;
      float bias = bq[col];
#pragma unroll
      for (int vv = 0; vv < 8; ++vv)
        q_sh[(r * 16 + mbase + vv) * 512 + col] = acc[j][vv] + bias;
    }
  }
  __syncthreads();

  v8f oacc[8];
#pragma unroll
  for (int j = 0; j < 8; ++j) oacc[j] = vzero8();

  for (int h = 0; h < 8; ++h) {
    // ---- Phase 2: k_h, v_h from pre-converted LDS A-frags ------------------
    {
      v8f ka = vzero8(), va = vzero8();
      for (int kc = 0; kc < 16; ++kc) {
        const __bf16* ap = ctx_sh + (kc * 4 + r) * 512 + lane * 16;
        v16bf ah = *(const v16bf*)ap;
        v16bf al = *(const v16bf*)(ap + 32768);
        const __bf16* kp = wkvp + (size_t)(kc * 64 + h * 4 + c)      * 1024 + lane * 16;
        const __bf16* vp = wkvp + (size_t)(kc * 64 + 32 + h * 4 + c) * 1024 + lane * 16;
        v16bf kbh = *(const v16bf*)kp, kbl = *(const v16bf*)(kp + 512);
        v16bf vbh = *(const v16bf*)vp, vbl = *(const v16bf*)(vp + 512);
        ka = wmma_bf16(ah, kbh, ka);
        ka = wmma_bf16(al, kbh, ka);
        ka = wmma_bf16(ah, kbl, ka);
        va = wmma_bf16(ah, vbh, va);
        va = wmma_bf16(al, vbh, va);
        va = wmma_bf16(ah, vbl, va);
      }
      const float bkk = bkv[h * 64 + c * 16 + ln];
      const float bvv = bkv[512 + h * 64 + c * 16 + ln];
#pragma unroll
      for (int vv = 0; vv < 8; ++vv) {
        int off   = (r * 16 + mbase + vv) * 64 + c * 16 + ln;
        k_sh[off] = ka[vv] + bkk;
        v_sh[off] = va[vv] + bvv;
      }
    }
    __syncthreads();

    // ---- Phase 3: masked-softmax attention (4 tokens/batch, dh=64) --------
    {
      const int row = tid >> 3;     // 0..63  (= local batch*4 + token i)
      const int g   = tid & 7;      // dim group: 8 dims each
      const int i   = row & 3;
      const int br0 = row & ~3;
      const float* qrow = q_sh + row * 512 + h * 64 + g * 8;
      float qv[8];
#pragma unroll
      for (int d = 0; d < 8; ++d) qv[d] = qrow[d];

      float s[4];
#pragma unroll
      for (int j = 0; j < 4; ++j) {
        const float* kr = k_sh + (br0 + j) * 64 + g * 8;
        float acc = 0.f;
#pragma unroll
        for (int d = 0; d < 8; ++d) acc += qv[d] * kr[d];
        s[j] = acc;
      }
      // reduce over the 8 lanes sharing this row
#pragma unroll
      for (int msk = 1; msk < 8; msk <<= 1) {
#pragma unroll
        for (int j = 0; j < 4; ++j) s[j] += __shfl_xor(s[j], msk, 32);
      }
      const int jm = 3 - i;  // the one masked position (dist > 1)
      float lg[4];
#pragma unroll
      for (int j = 0; j < 4; ++j)
        lg[j] = (j == jm) ? (-(float)(jm + 1) * kGamma) : s[j] * kScale;
      float mx = fmaxf(fmaxf(lg[0], lg[1]), fmaxf(lg[2], lg[3]));
      float ex[4], sum = 0.f;
#pragma unroll
      for (int j = 0; j < 4; ++j) {
        ex[j] = expf(lg[j] - mx);
        sum += ex[j];
      }
      float inv = 1.0f / sum;
      float p[4];
#pragma unroll
      for (int j = 0; j < 4; ++j) p[j] = (j == jm) ? 0.f : ex[j] * inv;
#pragma unroll
      for (int d = 0; d < 8; ++d) {
        float o = 0.f;
#pragma unroll
        for (int j = 0; j < 4; ++j) o += p[j] * v_sh[(br0 + j) * 64 + g * 8 + d];
        o_sh[row * 64 + g * 8 + d] = o;
      }
    }
    __syncthreads();

    // ---- Phase 4: out_acc += o_h @ Wo[h*64 .. h*64+63, :] ------------------
    {
      const float* orow = o_sh + (r * 16 + ln) * 64;
      for (int kk = 0; kk < 2; ++kk) {
        float av[16];
        v16bf ah, al;
        load_a16(orow, kk * 32 + kbase, av);
        cvt_split16(av, ah, al);
        const int kcg = h * 2 + kk;
#pragma unroll
        for (int j = 0; j < 8; ++j) {
          const __bf16* bp = wop + (size_t)(kcg * 32 + c * 8 + j) * 1024 + lane * 16;
          v16bf bh = *(const v16bf*)bp;
          v16bf bl = *(const v16bf*)(bp + 512);
          oacc[j] = wmma_bf16(ah, bh, oacc[j]);
          oacc[j] = wmma_bf16(al, bh, oacc[j]);
          oacc[j] = wmma_bf16(ah, bl, oacc[j]);
        }
      }
    }
    // No barrier needed here: the next Phase 2's post-barrier separates these
    // o_sh reads from the next Phase 3's o_sh writes, and k/v writes in the
    // next Phase 2 only begin after every thread passed the post-Phase-3
    // barrier of this iteration.
  }

  // ---- Epilogue: + bo, store 64x512 tile ----------------------------------
#pragma unroll
  for (int j = 0; j < 8; ++j) {
    int col    = c * 128 + j * 16 + ln;
    float bias = bo[col];
#pragma unroll
    for (int vv = 0; vv < 8; ++vv)
      out[(size_t)(row0 + r * 16 + mbase + vv) * 512 + col] = oacc[j][vv] + bias;
  }
}

// ---------------------------------------------------------------------------
// Host launcher.  d_ws layout (bf16 elements):
//   [0 .. 524288)          Wq  packed   (1 MB)
//   [524288 .. 1572864)    Wkv packed   (2 MB)
//   [1572864 .. 2097152)   Wo  packed   (1 MB)
// ---------------------------------------------------------------------------
extern "C" void kernel_launch(void* const* d_in, const int* in_sizes, int n_in,
                              void* d_out, int out_size, void* d_ws, size_t ws_size,
                              hipStream_t stream) {
  (void)in_sizes; (void)n_in; (void)out_size; (void)ws_size;
  const float* x    = (const float*)d_in[0];
  const float* ctx  = (const float*)d_in[1];
  const float* Wq   = (const float*)d_in[2];
  const float* bq   = (const float*)d_in[3];
  const float* Wkv  = (const float*)d_in[4];
  const float* bkv  = (const float*)d_in[5];
  const float* Wo   = (const float*)d_in[6];
  const float* bo   = (const float*)d_in[7];
  float* out        = (float*)d_out;

  __bf16* wqp  = (__bf16*)d_ws;
  __bf16* wkvp = wqp + 524288;
  __bf16* wop  = wkvp + 1048576;

  pack_weights_kernel<<<262144 / 256, 256, 0, stream>>>(Wq, 512, 512, wqp);
  pack_weights_kernel<<<524288 / 256, 256, 0, stream>>>(Wkv, 512, 1024, wkvp);
  pack_weights_kernel<<<262144 / 256, 256, 0, stream>>>(Wo, 512, 512, wop);

  fused_attn_kernel<<<1024, 512, 0, stream>>>(x, ctx, bq, bkv, bo,
                                              wqp, wkvp, wop, out);
}
